// MultiHeadAttention_8383776161935
// MI455X (gfx1250) — matrix operand order repaired and checked by reading, not compiled, not run on hardware
//
#include <hip/hip_runtime.h>
#include <hip/hip_bf16.h>

typedef __bf16 bf16_t;
typedef __attribute__((ext_vector_type(16))) __bf16 v16bf;
typedef __attribute__((ext_vector_type(8)))  float  v8f;
typedef __attribute__((ext_vector_type(4)))  int    v4i;
typedef __attribute__((ext_vector_type(8)))  int    v8i_t;

#define D_MODEL 1024
#define HEADS   16
#define DEPTH   64
#define BATCH   4
#define SEQ     2048
#define MTOT    (BATCH * SEQ)   // 8192

#define APITCH 40   // bf16 elements per LDS A row (32 used + 8 pad) = 80B, 16B aligned
#define BPITCH 40

// ---------------------------------------------------------------------------
// WMMA helper: D = A(16x32 bf16) * B(32x16 bf16) + C(f32)
// ---------------------------------------------------------------------------
__device__ __forceinline__ v8f wmma_bf16(v16bf a, v16bf b, v8f c) {
  return __builtin_amdgcn_wmma_f32_16x16x32_bf16(false, a, false, b,
                                                 (short)0, c, false, false);
}

__device__ __forceinline__ void wait_tensorcnt0() {
#if __has_builtin(__builtin_amdgcn_s_wait_tensorcnt)
  __builtin_amdgcn_s_wait_tensorcnt(0);
#else
  asm volatile("s_wait_tensorcnt 0x0" ::: "memory");
#endif
}

// ---------------------------------------------------------------------------
// Tensor Data Mover: DMA a 2D bf16 tile (tile_d0 elems/row x tile_d1 rows,
// source row stride stride0 elems) from global into LDS, inserting 4 DWORDs
// of padding after every 16 DWORDs (row = 32 bf16 = 64B) so LDS rows land on
// the 40-element (80B) pitch used by the WMMA fragment readers.
// D# layout per CDNA5 ISA sec.8. Wave-level op: call from
// ONE wave only; EXEC is ignored by TDM.
// ---------------------------------------------------------------------------
__device__ __forceinline__ void tdm_load_2d_bf16(const void* gptr, void* lptr,
                                                 unsigned tensor_d0, unsigned tensor_d1,
                                                 unsigned tile_d0, unsigned tile_d1,
                                                 unsigned stride0_elems) {
  const unsigned long long ga = (unsigned long long)(uintptr_t)gptr;
  const unsigned lds = (unsigned)(uintptr_t)lptr;  // low 32 bits = LDS offset

  v4i g0;
  g0[0] = 1;                                   // count=1, user mode, no gather
  g0[1] = (int)lds;                            // lds_addr
  g0[2] = (int)(ga & 0xffffffffull);           // global_addr[31:0]
  g0[3] = (int)((unsigned)(ga >> 32) | (2u << 30));  // global_addr[56:32] | type=2

  v8i_t g1;
  // workgroup_mask=0 | data_size=1(2B)<<16 | pad_enable<<20 |
  // pad_interval=3(16 DWORDs)<<22 | pad_amount=3(4 DWORDs)<<25
  g1[0] = (1 << 16) | (1 << 20) | (3 << 22) | (3 << 25);
  g1[1] = (int)((tensor_d0 & 0xffffu) << 16);                    // tensor_dim0[15:0]
  g1[2] = (int)((tensor_d0 >> 16) | ((tensor_d1 & 0xffffu) << 16)); // td0 hi | td1 lo
  g1[3] = (int)((tensor_d1 >> 16) | (tile_d0 << 16));            // td1 hi | tile_dim0
  g1[4] = (int)(tile_d1 & 0xffffu);                              // tile_dim1 (tile_dim2=0)
  g1[5] = (int)stride0_elems;                                    // tensor_dim0_stride lo
  g1[6] = 0;                                                     // stride0 hi | stride1 lo
  g1[7] = 0;                                                     // stride1 hi

  asm volatile("tensor_load_to_lds %0, %1" :: "s"(g0), "s"(g1) : "memory");
}

// A fragment (16x32): lane<16 -> M=lane, K=0..15; lane>=16 -> K=16..31
__device__ __forceinline__ v16bf load_a_frag(const bf16_t* As, int row_base, int lane) {
  v16bf a;
  const int m  = lane & 15;
  const int kb = (lane >> 4) * 16;
  const bf16_t* rowp = As + (size_t)(row_base + m) * APITCH;
#pragma unroll
  for (int i = 0; i < 16; ++i) {
    const int k = kb + i;
    a[i] = rowp[k];
  }
  return a;
}

// B fragment (32x16): Bs stored [n][k]; lane<16 -> N=lane, K=0..15; lane>=16 -> K=16..31
__device__ __forceinline__ v16bf load_b_frag(const bf16_t* Bs, int col_base, int lane) {
  v16bf b;
  const int n  = lane & 15;
  const int kb = (lane >> 4) * 16;
  const bf16_t* colp = Bs + (size_t)(col_base + n) * BPITCH;
#pragma unroll
  for (int i = 0; i < 16; ++i) b[i] = colp[kb + i];
  return b;
}

// ---------------------------------------------------------------------------
// Tile loaders (256 threads). Tiles: A = 128 rows x 32 K, B = 32 K x 64 N.
// ---------------------------------------------------------------------------
__device__ __forceinline__ void load_tile_a_f32(const float* __restrict__ src,
                                                size_t row_stride, int t, bf16_t* As) {
  const int r = t >> 3;
  const int c = (t & 7) * 4;
#pragma unroll
  for (int p = 0; p < 4; ++p) {
    const int rr = r + p * 32;
    const float4 v = *(const float4*)(src + (size_t)rr * row_stride + c);
    bf16_t* dst = As + (size_t)rr * APITCH + c;
    dst[0] = (bf16_t)v.x; dst[1] = (bf16_t)v.y;
    dst[2] = (bf16_t)v.z; dst[3] = (bf16_t)v.w;
  }
}

// B from f32 with transpose: src at (k0, n0); Bs[n][k] = src[k][n]
__device__ __forceinline__ void load_tile_b_f32t(const float* __restrict__ src,
                                                 size_t row_stride, int t, bf16_t* Bs) {
  const int k  = t >> 3;          // 0..31
  const int n0 = (t & 7) * 8;     // 0..56
  const float4 v0 = *(const float4*)(src + (size_t)k * row_stride + n0);
  const float4 v1 = *(const float4*)(src + (size_t)k * row_stride + n0 + 4);
  Bs[(size_t)(n0 + 0) * BPITCH + k] = (bf16_t)v0.x;
  Bs[(size_t)(n0 + 1) * BPITCH + k] = (bf16_t)v0.y;
  Bs[(size_t)(n0 + 2) * BPITCH + k] = (bf16_t)v0.z;
  Bs[(size_t)(n0 + 3) * BPITCH + k] = (bf16_t)v0.w;
  Bs[(size_t)(n0 + 4) * BPITCH + k] = (bf16_t)v1.x;
  Bs[(size_t)(n0 + 5) * BPITCH + k] = (bf16_t)v1.y;
  Bs[(size_t)(n0 + 6) * BPITCH + k] = (bf16_t)v1.z;
  Bs[(size_t)(n0 + 7) * BPITCH + k] = (bf16_t)v1.w;
}

// B from bf16 with transpose: src at (k0, n0) row-major (k rows); Bs[n][k] = src[k][n]
__device__ __forceinline__ void load_tile_b_bf16t(const bf16_t* __restrict__ src,
                                                  size_t row_stride, int t, bf16_t* Bs) {
  const int k  = t >> 3;          // 0..31
  const int n0 = (t & 7) * 8;     // 0..56
  union { uint4 u; bf16_t h[8]; } tmp;
  tmp.u = *(const uint4*)(src + (size_t)k * row_stride + n0);
#pragma unroll
  for (int j = 0; j < 8; ++j) Bs[(size_t)(n0 + j) * BPITCH + k] = tmp.h[j];
}

// ---------------------------------------------------------------------------
// Kernel 1: projection  Yh(b,h,s,d) = bf16( X @ W + bias )   [X,W f32]
// grid: (MTOT/128, D_MODEL/64), 256 threads
// ---------------------------------------------------------------------------
__global__ __launch_bounds__(256) void proj_kernel(const float* __restrict__ X,
                                                   const float* __restrict__ W,
                                                   const float* __restrict__ bias,
                                                   bf16_t* __restrict__ Yh) {
  __shared__ bf16_t As[128 * APITCH];
  __shared__ bf16_t Bs[64 * BPITCH];

  const int t    = threadIdx.x;
  const int lane = t & 31;
  const int wave = t >> 5;
  const int wm   = wave >> 1;   // 0..3
  const int wn   = wave & 1;    // 0..1
  const int row0 = blockIdx.x * 128;
  const int col0 = blockIdx.y * 64;

  const v8f vzero = {0.f, 0.f, 0.f, 0.f, 0.f, 0.f, 0.f, 0.f};
  v8f acc[2][2];
#pragma unroll
  for (int i = 0; i < 2; ++i)
#pragma unroll
    for (int j = 0; j < 2; ++j) acc[i][j] = vzero;

  for (int kk = 0; kk < D_MODEL; kk += 32) {
    load_tile_a_f32(X + (size_t)row0 * D_MODEL + kk, D_MODEL, t, As);
    load_tile_b_f32t(W + (size_t)kk * D_MODEL + col0, D_MODEL, t, Bs);
    __syncthreads();
    if (kk + 32 < D_MODEL) {
      __builtin_prefetch(X + (size_t)(row0 + (t >> 3)) * D_MODEL + kk + 32, 0, 3);
      __builtin_prefetch(W + (size_t)(kk + 32 + (t >> 3)) * D_MODEL + col0, 0, 3);
    }
    const v16bf a0 = load_a_frag(As, wm * 32 + 0,  lane);
    const v16bf a1 = load_a_frag(As, wm * 32 + 16, lane);
    const v16bf b0 = load_b_frag(Bs, wn * 32 + 0,  lane);
    const v16bf b1 = load_b_frag(Bs, wn * 32 + 16, lane);
    acc[0][0] = wmma_bf16(a0, b0, acc[0][0]);
    acc[0][1] = wmma_bf16(a0, b1, acc[0][1]);
    acc[1][0] = wmma_bf16(a1, b0, acc[1][0]);
    acc[1][1] = wmma_bf16(a1, b1, acc[1][1]);
    __syncthreads();
  }

  const int cm = 8 * (lane >> 4);
  const int cn = lane & 15;
#pragma unroll
  for (int tm = 0; tm < 2; ++tm)
#pragma unroll
    for (int tn = 0; tn < 2; ++tn)
#pragma unroll
      for (int r = 0; r < 8; ++r) {
        const int grow = row0 + wm * 32 + tm * 16 + cm + r;  // token index
        const int gcol = col0 + wn * 32 + tn * 16 + cn;      // model dim
        const float val = acc[tm][tn][r] + bias[gcol];
        const int b = grow >> 11, s = grow & (SEQ - 1);
        const int h = gcol >> 6,  d = gcol & (DEPTH - 1);
        Yh[(((size_t)(b * HEADS + h)) * SEQ + s) * DEPTH + d] = (bf16_t)val;
      }
}

// ---------------------------------------------------------------------------
// Kernel 2: logits = (Qh @ Kh^T) * 1/sqrt(D) + mask-bias   (pre-softmax, f32)
// A and B tiles staged by the Tensor Data Mover (wave 0 issues the DMA).
// grid: (SEQ/128, SEQ/64, BATCH*HEADS)
// ---------------------------------------------------------------------------
__global__ __launch_bounds__(256) void logits_kernel(const bf16_t* __restrict__ Qh,
                                                     const bf16_t* __restrict__ Kh,
                                                     const unsigned char* __restrict__ mask,
                                                     float* __restrict__ attn) {
  __shared__ bf16_t As[128 * APITCH];
  __shared__ bf16_t Bs[64 * BPITCH];

  const int t    = threadIdx.x;
  const int lane = t & 31;
  const int wave = t >> 5;
  const int wm   = wave >> 1;
  const int wn   = wave & 1;
  const int bh   = blockIdx.z;
  const int b    = bh / HEADS;
  const int row0 = blockIdx.x * 128;   // query position
  const int col0 = blockIdx.y * 64;    // key position
  const size_t base = (size_t)bh * SEQ * DEPTH;

  const v8f vzero = {0.f, 0.f, 0.f, 0.f, 0.f, 0.f, 0.f, 0.f};
  v8f acc[2][2];
#pragma unroll
  for (int i = 0; i < 2; ++i)
#pragma unroll
    for (int j = 0; j < 2; ++j) acc[i][j] = vzero;

#pragma unroll
  for (int kk = 0; kk < DEPTH; kk += 32) {
    if (wave == 0) {
      // A: 128 q-rows x 32 d, row stride DEPTH; B: 64 k-rows x 32 d.
      tdm_load_2d_bf16(Qh + base + (size_t)row0 * DEPTH + kk, As,
                       DEPTH, SEQ, 32, 128, DEPTH);
      tdm_load_2d_bf16(Kh + base + (size_t)col0 * DEPTH + kk, Bs,
                       DEPTH, SEQ, 32, 64, DEPTH);
      wait_tensorcnt0();
    }
    __syncthreads();
    const v16bf a0 = load_a_frag(As, wm * 32 + 0,  lane);
    const v16bf a1 = load_a_frag(As, wm * 32 + 16, lane);
    const v16bf b0 = load_b_frag(Bs, wn * 32 + 0,  lane);
    const v16bf b1 = load_b_frag(Bs, wn * 32 + 16, lane);
    acc[0][0] = wmma_bf16(a0, b0, acc[0][0]);
    acc[0][1] = wmma_bf16(a0, b1, acc[0][1]);
    acc[1][0] = wmma_bf16(a1, b0, acc[1][0]);
    acc[1][1] = wmma_bf16(a1, b1, acc[1][1]);
    __syncthreads();
  }

  const float scale = 0.125f;  // 1/sqrt(64)
  const int cm = 8 * (lane >> 4);
  const int cn = lane & 15;
#pragma unroll
  for (int tm = 0; tm < 2; ++tm)
#pragma unroll
    for (int tn = 0; tn < 2; ++tn)
#pragma unroll
      for (int r = 0; r < 8; ++r) {
        const int q = row0 + wm * 32 + tm * 16 + cm + r;
        const int kp = col0 + wn * 32 + tn * 16 + cn;
        const size_t mi = ((size_t)b * SEQ + q) * SEQ + kp;
        const float mb = mask[mi] ? 0.0f : -1.0e9f;
        attn[((size_t)bh * SEQ + q) * SEQ + kp] = acc[tm][tn][r] * scale + mb;
      }
}

// ---------------------------------------------------------------------------
// Kernel 3: in-place row softmax over SK=2048. One block per row.
// ---------------------------------------------------------------------------
__global__ __launch_bounds__(256) void softmax_kernel(float* __restrict__ attn) {
  float* p = attn + (size_t)blockIdx.x * SEQ;
  const int t = threadIdx.x;
  const int lane = t & 31, wave = t >> 5;
  __shared__ float red[8];

  float m = -3.0e38f;
  for (int i = t; i < SEQ; i += 256) m = fmaxf(m, p[i]);
#pragma unroll
  for (int off = 16; off > 0; off >>= 1) m = fmaxf(m, __shfl_xor(m, off, 32));
  if (lane == 0) red[wave] = m;
  __syncthreads();
  float rm = red[0];
#pragma unroll
  for (int i = 1; i < 8; ++i) rm = fmaxf(rm, red[i]);
  __syncthreads();

  float s = 0.f;
  for (int i = t; i < SEQ; i += 256) {
    const float e = __expf(p[i] - rm);
    p[i] = e;
    s += e;
  }
#pragma unroll
  for (int off = 16; off > 0; off >>= 1) s += __shfl_xor(s, off, 32);
  if (lane == 0) red[wave] = s;
  __syncthreads();
  float rs = 0.f;
#pragma unroll
  for (int i = 0; i < 8; ++i) rs += red[i];
  const float inv = 1.0f / rs;
  for (int i = t; i < SEQ; i += 256) p[i] *= inv;
}

// ---------------------------------------------------------------------------
// Kernel 4: ctx(b, s, h*64+d) = bf16( attn(bh) @ Vh(bh) )
// grid: (SEQ/128, 1, BATCH*HEADS)
// ---------------------------------------------------------------------------
__global__ __launch_bounds__(256) void ctx_kernel(const float* __restrict__ attn,
                                                  const bf16_t* __restrict__ Vh,
                                                  bf16_t* __restrict__ Ctx) {
  __shared__ bf16_t As[128 * APITCH];
  __shared__ bf16_t Bs[64 * BPITCH];

  const int t    = threadIdx.x;
  const int lane = t & 31;
  const int wave = t >> 5;
  const int wm   = wave >> 1;
  const int wn   = wave & 1;
  const int bh   = blockIdx.z;
  const int b    = bh / HEADS;
  const int h    = bh % HEADS;
  const int row0 = blockIdx.x * 128;
  const size_t abase = (size_t)bh * SEQ * SEQ;
  const size_t vbase = (size_t)bh * SEQ * DEPTH;

  const v8f vzero = {0.f, 0.f, 0.f, 0.f, 0.f, 0.f, 0.f, 0.f};
  v8f acc[2][2];
#pragma unroll
  for (int i = 0; i < 2; ++i)
#pragma unroll
    for (int j = 0; j < 2; ++j) acc[i][j] = vzero;

  for (int kk = 0; kk < SEQ; kk += 32) {
    load_tile_a_f32(attn + abase + (size_t)row0 * SEQ + kk, SEQ, t, As);
    load_tile_b_bf16t(Vh + vbase + (size_t)kk * DEPTH, DEPTH, t, Bs);
    __syncthreads();
    if (kk + 32 < SEQ) {
      __builtin_prefetch(attn + abase + (size_t)(row0 + (t >> 3)) * SEQ + kk + 32, 0, 3);
    }
    const v16bf a0 = load_a_frag(As, wm * 32 + 0,  lane);
    const v16bf a1 = load_a_frag(As, wm * 32 + 16, lane);
    const v16bf b0 = load_b_frag(Bs, wn * 32 + 0,  lane);
    const v16bf b1 = load_b_frag(Bs, wn * 32 + 16, lane);
    acc[0][0] = wmma_bf16(a0, b0, acc[0][0]);
    acc[0][1] = wmma_bf16(a0, b1, acc[0][1]);
    acc[1][0] = wmma_bf16(a1, b0, acc[1][0]);
    acc[1][1] = wmma_bf16(a1, b1, acc[1][1]);
    __syncthreads();
  }

  const int cm = 8 * (lane >> 4);
  const int cn = lane & 15;
#pragma unroll
  for (int tm = 0; tm < 2; ++tm)
#pragma unroll
    for (int tn = 0; tn < 2; ++tn)
#pragma unroll
      for (int r = 0; r < 8; ++r) {
        const int s = row0 + wm * 32 + tm * 16 + cm + r;
        const int d = wn * 32 + tn * 16 + cn;          // 0..63
        Ctx[((size_t)b * SEQ + s) * D_MODEL + h * DEPTH + d] = (bf16_t)acc[tm][tn][r];
      }
}

// ---------------------------------------------------------------------------
// Kernel 5: out = Ctx(bf16) @ Wo(f32) + bias  -> f32
// A tile staged by the Tensor Data Mover.
// grid: (MTOT/128, D_MODEL/64)
// ---------------------------------------------------------------------------
__global__ __launch_bounds__(256) void out_kernel(const bf16_t* __restrict__ Ctx,
                                                  const float* __restrict__ Wo,
                                                  const float* __restrict__ bias,
                                                  float* __restrict__ Out) {
  __shared__ bf16_t As[128 * APITCH];
  __shared__ bf16_t Bs[64 * BPITCH];

  const int t    = threadIdx.x;
  const int lane = t & 31;
  const int wave = t >> 5;
  const int wm   = wave >> 1;
  const int wn   = wave & 1;
  const int row0 = blockIdx.x * 128;
  const int col0 = blockIdx.y * 64;

  const v8f vzero = {0.f, 0.f, 0.f, 0.f, 0.f, 0.f, 0.f, 0.f};
  v8f acc[2][2];
#pragma unroll
  for (int i = 0; i < 2; ++i)
#pragma unroll
    for (int j = 0; j < 2; ++j) acc[i][j] = vzero;

  for (int kk = 0; kk < D_MODEL; kk += 32) {
    if (wave == 0) {
      tdm_load_2d_bf16(Ctx + (size_t)row0 * D_MODEL + kk, As,
                       D_MODEL, MTOT, 32, 128, D_MODEL);
    }
    load_tile_b_f32t(Wo + (size_t)kk * D_MODEL + col0, D_MODEL, t, Bs);
    if (wave == 0) wait_tensorcnt0();
    __syncthreads();
    if (kk + 32 < D_MODEL) {
      __builtin_prefetch(Wo + (size_t)(kk + 32 + (t >> 3)) * D_MODEL + col0, 0, 3);
    }
    const v16bf a0 = load_a_frag(As, wm * 32 + 0,  lane);
    const v16bf a1 = load_a_frag(As, wm * 32 + 16, lane);
    const v16bf b0 = load_b_frag(Bs, wn * 32 + 0,  lane);
    const v16bf b1 = load_b_frag(Bs, wn * 32 + 16, lane);
    acc[0][0] = wmma_bf16(a0, b0, acc[0][0]);
    acc[0][1] = wmma_bf16(a0, b1, acc[0][1]);
    acc[1][0] = wmma_bf16(a1, b0, acc[1][0]);
    acc[1][1] = wmma_bf16(a1, b1, acc[1][1]);
    __syncthreads();
  }

  const int cm = 8 * (lane >> 4);
  const int cn = lane & 15;
#pragma unroll
  for (int tm = 0; tm < 2; ++tm)
#pragma unroll
    for (int tn = 0; tn < 2; ++tn)
#pragma unroll
      for (int r = 0; r < 8; ++r) {
        const int grow = row0 + wm * 32 + tm * 16 + cm + r;
        const int gcol = col0 + wn * 32 + tn * 16 + cn;
        Out[(size_t)grow * D_MODEL + gcol] = acc[tm][tn][r] + bias[gcol];
      }
}

// ---------------------------------------------------------------------------
// Host-side launcher
// ---------------------------------------------------------------------------
extern "C" void kernel_launch(void* const* d_in, const int* in_sizes, int n_in,
                              void* d_out, int out_size, void* d_ws, size_t ws_size,
                              hipStream_t stream) {
  (void)in_sizes; (void)n_in; (void)out_size; (void)ws_size;

  const float* q    = (const float*)d_in[0];
  const float* k    = (const float*)d_in[1];
  const float* v    = (const float*)d_in[2];
  const unsigned char* mask = (const unsigned char*)d_in[3];
  const float* wq_w = (const float*)d_in[4];
  const float* wq_b = (const float*)d_in[5];
  const float* wk_w = (const float*)d_in[6];
  const float* wk_b = (const float*)d_in[7];
  const float* wv_w = (const float*)d_in[8];
  const float* wv_b = (const float*)d_in[9];
  const float* wo_w = (const float*)d_in[10];
  const float* wo_b = (const float*)d_in[11];

  float* out  = (float*)d_out;                       // (B, SQ, D_MODEL)
  float* attn = out + (size_t)MTOT * D_MODEL;        // (B, H, SQ, SK)

  bf16_t* qh  = (bf16_t*)d_ws;                       // (B,H,SEQ,DEPTH) bf16
  bf16_t* kh  = qh + (size_t)MTOT * D_MODEL;
  bf16_t* vh  = kh + (size_t)MTOT * D_MODEL;
  bf16_t* ctx = vh + (size_t)MTOT * D_MODEL;         // (B,SEQ,D_MODEL) bf16

  dim3 pgrid(MTOT / 128, D_MODEL / 64);
  proj_kernel<<<pgrid, 256, 0, stream>>>(q, wq_w, wq_b, qh);
  proj_kernel<<<pgrid, 256, 0, stream>>>(k, wk_w, wk_b, kh);
  proj_kernel<<<pgrid, 256, 0, stream>>>(v, wv_w, wv_b, vh);

  dim3 lgrid(SEQ / 128, SEQ / 64, BATCH * HEADS);
  logits_kernel<<<lgrid, 256, 0, stream>>>(qh, kh, mask, attn);

  softmax_kernel<<<dim3(BATCH * HEADS * SEQ), 256, 0, stream>>>(attn);

  dim3 cgrid(SEQ / 128, 1, BATCH * HEADS);
  ctx_kernel<<<cgrid, 256, 0, stream>>>(attn, vh, ctx);

  dim3 ogrid(MTOT / 128, D_MODEL / 64);
  out_kernel<<<ogrid, 256, 0, stream>>>(ctx, wo_w, wo_b, out);
}
